// TitansL2_46377056862423
// MI455X (gfx1250) — matrix operand order repaired and checked by reading, not compiled, not run on hardware
//
#include <hip/hip_runtime.h>
#include <stdint.h>
#include <math.h>

// ---------------------------------------------------------------------------
// CDNA5 (gfx1250, wave32) types
// ---------------------------------------------------------------------------
typedef __attribute__((ext_vector_type(16))) _Float16 v16h;
typedef __attribute__((ext_vector_type(8)))  _Float16 h8;
typedef __attribute__((ext_vector_type(4)))  _Float16 h4;
typedef __attribute__((ext_vector_type(8)))  float    v8f;

union FragU { h8 u[2]; v16h v; };

#define TILE_M 128
#define TILE_N 128
#define TILE_K 32
#define LDA    40   // padded LDS row stride (halves): 80B rows -> conflict-free b128

// ---------------------------------------------------------------------------
// Tensor Data Mover path (gfx1250 TDM). Descriptor per CDNA5 ISA:
//  group0: count=1 | lds_addr | global_addr | type=2
//  group1: data_size=2B, pad_enable, pad_interval(code 3 => 16 DWORDs = one
//          64B row), pad_amount(code 3 => 4 DWORDs = 16B)  ==> LDS row stride
//          80B, matching LDA=40 halves. tensor_dim0_stride = K.
// ---------------------------------------------------------------------------
#if defined(__HIP_DEVICE_COMPILE__) && __has_builtin(__builtin_amdgcn_tensor_load_to_lds)
#define USE_TDM 1
#else
#define USE_TDM 0
#endif

#if USE_TDM
typedef __attribute__((ext_vector_type(4))) unsigned int u32x4;
typedef __attribute__((ext_vector_type(8))) int          i32x8;
typedef __attribute__((ext_vector_type(4))) int          i32x4;

__device__ __forceinline__ void tdm_load_tile_f16(
    const _Float16* gsrc, unsigned lds_byte_addr,
    unsigned tile_w_elems, unsigned tile_h_rows, unsigned row_stride_elems,
    unsigned pad_interval_code, unsigned pad_amount_code)
{
  const unsigned long long ga = (unsigned long long)(uintptr_t)gsrc;
  u32x4 g0;
  g0.x = 1u;                                                // count=1 (valid)
  g0.y = lds_byte_addr;                                     // LDS dest (bytes)
  g0.z = (unsigned)ga;                                      // global addr lo
  g0.w = (unsigned)((ga >> 32) & 0x01FFFFFFu) | (2u << 30); // addr hi | type=2

  union { unsigned long long q[4]; i32x8 v; } g1;
  const unsigned long long dim0 = row_stride_elems;  // tensor_dim0 (elems)
  const unsigned long long dim1 = 1ull << 20;        // tensor_dim1 (ample)
  const unsigned long long s0   = row_stride_elems;  // tensor_dim0_stride
  g1.q[0] = (1ull << 16)                             // data_size = 2 bytes
          | (1ull << 20)                             // pad_enable
          | ((unsigned long long)pad_interval_code << 22)
          | ((unsigned long long)pad_amount_code  << 25)
          | ((dim0 & 0xFFFFull) << 48);
  g1.q[1] = ((dim0 >> 16) & 0xFFFFull)
          | ((dim1 & 0xFFFFFFFFull) << 16)
          | ((unsigned long long)(tile_w_elems & 0xFFFFu) << 48); // tile_dim0
  g1.q[2] = (unsigned long long)(tile_h_rows & 0xFFFFu)           // tile_dim1
          | ((s0 & 0xFFFFFFFFull) << 32);
  g1.q[3] = (s0 >> 32) & 0xFFFFull;

  const i32x4 z4 = {0, 0, 0, 0};
#if __clang_major__ >= 23
  const i32x8 z8 = {0, 0, 0, 0, 0, 0, 0, 0};
  __builtin_amdgcn_tensor_load_to_lds(g0, g1.v, z4, z4, z8, 0);
#else
  __builtin_amdgcn_tensor_load_to_lds(g0, g1.v, z4, z4, 0);
#endif
}
#endif // USE_TDM

// ---------------------------------------------------------------------------
// C[M,N] = A[M,K] @ Bt[N,K]^T, all-f16 operands (pre-converted), f32 accum.
// 256 threads = 8 waves; block tile 128x128; wave owns 32x64 (2x4 WMMA frags).
// TDM path: double-buffered LDS tiles filled by tensor_load_to_lds with LDS
// padding; s_wait_tensorcnt(2) overlaps current compute with next-tile DMA.
// ---------------------------------------------------------------------------
__global__ __launch_bounds__(256)
void gemm_f16_wmma(const _Float16* __restrict__ A,   // [M,K] row-major
                   const _Float16* __restrict__ Bt,  // [N,K] row-major (W^T)
                   float* __restrict__ C, int M, int N, int K)
{
  __shared__ _Float16 As[2][TILE_M][LDA];
  __shared__ _Float16 Bs[2][TILE_N][LDA];

  const int t    = threadIdx.x;
  const int lane = t & 31;
  const int wave = t >> 5;
  const int wr   = wave & 3;            // rows wr*32
  const int wc   = wave >> 2;           // cols wc*64
  const int rowBase = blockIdx.x * TILE_M;
  const int colBase = blockIdx.y * TILE_N;
  const int m16 = lane & 15;
  const int hh  = lane >> 4;

  v8f acc[2][4];
  const v8f vz = {0.f, 0.f, 0.f, 0.f, 0.f, 0.f, 0.f, 0.f};
#pragma unroll
  for (int rt = 0; rt < 2; ++rt)
#pragma unroll
    for (int ct = 0; ct < 4; ++ct) acc[rt][ct] = vz;

#if USE_TDM
  if (wave == 0) {
    tdm_load_tile_f16(A  + (size_t)rowBase * K,
                      (unsigned)(uintptr_t)&As[0][0][0], TILE_K, TILE_M, K, 3, 3);
    tdm_load_tile_f16(Bt + (size_t)colBase * K,
                      (unsigned)(uintptr_t)&Bs[0][0][0], TILE_K, TILE_N, K, 3, 3);
  }
#endif

  int buf = 0;
  for (int k0 = 0; k0 < K; k0 += TILE_K) {
#if USE_TDM
    const bool last = (k0 + TILE_K >= K);
    if (wave == 0 && !last) {           // prefetch next tile into other buffer
      tdm_load_tile_f16(A  + (size_t)rowBase * K + (k0 + TILE_K),
                        (unsigned)(uintptr_t)&As[buf ^ 1][0][0],
                        TILE_K, TILE_M, K, 3, 3);
      tdm_load_tile_f16(Bt + (size_t)colBase * K + (k0 + TILE_K),
                        (unsigned)(uintptr_t)&Bs[buf ^ 1][0][0],
                        TILE_K, TILE_N, K, 3, 3);
    }
    // TDM ops complete in order: <=2 outstanding  => current tile landed.
    if (last) __builtin_amdgcn_s_wait_tensorcnt(0);
    else      __builtin_amdgcn_s_wait_tensorcnt(2);
    __syncthreads();
#else
    // Cooperative f16 staging fallback (no conversion: operands already f16).
    if ((k0 + TILE_K) < K && t == 0) {
      __builtin_prefetch(A  + (size_t)rowBase * K + (k0 + TILE_K), 0, 1);
      __builtin_prefetch(Bt + (size_t)colBase * K + (k0 + TILE_K), 0, 1);
    }
#pragma unroll
    for (int i = 0; i < 2; ++i) {
      const int idx = t + i * 256;      // 0..511
      const int r   = idx >> 2;         // 0..127
      const int c8  = idx & 3;          // h8 chunk
      *(h8*)&As[0][r][c8 * 8] =
          *(const h8*)(A + (size_t)(rowBase + r) * K + k0 + c8 * 8);
      *(h8*)&Bs[0][r][c8 * 8] =
          *(const h8*)(Bt + (size_t)(colBase + r) * K + k0 + c8 * 8);
    }
    __syncthreads();
#endif

    // Fragments per wave32 16-bit WMMA layouts (CDNA5 ISA):
    // A 16x32: lane = row m16; K runs {8hh..+8} and {16+8hh..+8}
    // B 32x16 (from transposed LDS): lane = col m16; K runs {8hh..+8} and {16+8hh..+8}
    FragU afr[2];
#pragma unroll
    for (int rt = 0; rt < 2; ++rt) {
      const int r = wr * 32 + rt * 16 + m16;
      afr[rt].u[0] = *(const h8*)&As[buf][r][8 * hh];
      afr[rt].u[1] = *(const h8*)&As[buf][r][16 + 8 * hh];
    }
    FragU bfr[4];
#pragma unroll
    for (int ct = 0; ct < 4; ++ct) {
      const int n = wc * 64 + ct * 16 + m16;
      bfr[ct].u[0] = *(const h8*)&Bs[buf][n][8 * hh];
      bfr[ct].u[1] = *(const h8*)&Bs[buf][n][16 + 8 * hh];
    }

#pragma unroll
    for (int rt = 0; rt < 2; ++rt)
#pragma unroll
      for (int ct = 0; ct < 4; ++ct)
        acc[rt][ct] = __builtin_amdgcn_wmma_f32_16x16x32_f16(
            false, afr[rt].v, false, bfr[ct].v, (short)0, acc[rt][ct],
            false, false);
    __syncthreads();
#if USE_TDM
    buf ^= 1;
#endif
  }

  // D layout: VGPR r -> M = r (lanes 0-15) / 8+r (lanes 16-31), N = m16
#pragma unroll
  for (int rt = 0; rt < 2; ++rt)
#pragma unroll
    for (int ct = 0; ct < 4; ++ct)
#pragma unroll
      for (int r = 0; r < 8; ++r) {
        const int row = rowBase + wr * 32 + rt * 16 + hh * 8 + r;
        const int col = colBase + wc * 64 + ct * 16 + m16;
        C[(size_t)row * N + col] = acc[rt][ct][r];
      }
}

// ---------------------------------------------------------------------------
// f32 -> f16 vectorized conversion (x and y_pre operand precasting).
// ---------------------------------------------------------------------------
__global__ __launch_bounds__(256)
void f32_to_f16_vec(const float* __restrict__ in, _Float16* __restrict__ out,
                    int n4)
{
  const int i = blockIdx.x * blockDim.x + threadIdx.x;
  if (i < n4) {
    const float4 f = ((const float4*)in)[i];
    h4 hv = { (_Float16)f.x, (_Float16)f.y, (_Float16)f.z, (_Float16)f.w };
    ((h4*)out)[i] = hv;
  }
}

// ---------------------------------------------------------------------------
// W[K,N] f32 -> Wt[N,K] f16 (tiled transpose so GEMM B-staging is contiguous).
// ---------------------------------------------------------------------------
__global__ __launch_bounds__(256)
void transpose_f32_to_f16(const float* __restrict__ W,
                          _Float16* __restrict__ Wt, int Kdim, int Ndim)
{
  __shared__ _Float16 tile[32][33];
  const int tx = threadIdx.x;           // 0..31
  const int ty = threadIdx.y;           // 0..7
  const int n0 = blockIdx.x * 32;
  const int k0 = blockIdx.y * 32;
#pragma unroll
  for (int r = 0; r < 4; ++r) {
    const int k = k0 + ty + r * 8;
    tile[ty + r * 8][tx] = (_Float16)W[(size_t)k * Ndim + n0 + tx];
  }
  __syncthreads();
#pragma unroll
  for (int r = 0; r < 4; ++r) {
    const int n = n0 + ty + r * 8;
    Wt[(size_t)n * Kdim + k0 + tx] = tile[tx][ty + r * 8];
  }
}

// ---------------------------------------------------------------------------
// Direct Titans-L2 recurrence (one block per (b,h)):
//   y_t = M q_t (pre-update);  M <- M (I - a kn kn^T) + b v kn^T
// State in registers: thread t owns row i=t/4, cols [(t%4)*16,+16).
// y overwrites q in place; it is cast to f16 (into xh) before the Wo GEMM.
// ---------------------------------------------------------------------------
__global__ __launch_bounds__(256)
void titans_recurrence(float* __restrict__ q,
                       const float* __restrict__ k,
                       const float* __restrict__ v,
                       const float* __restrict__ alpha_raw,
                       const float* __restrict__ beta_raw,
                       float* __restrict__ M_out,    // [B,H,D,D]
                       int Bsz, int T, int H, int D)
{
  const int t = threadIdx.x;
  const int b = blockIdx.x / H;
  const int h = blockIdx.x % H;
  const int Cc = H * D;

  const float acoef = 0.5f / (1.f + __expf(-alpha_raw[h]));
  const float bcoef = 0.5f / (1.f + __expf(-beta_raw[h]));

  __shared__ float qs[64], ks[64], vs[64];
  __shared__ float s_rinv;

  const int i  = t >> 2;
  const int jb = (t & 3) * 16;

  float m[16];
#pragma unroll
  for (int jj = 0; jj < 16; ++jj) m[jj] = 0.f;

  for (int step = 0; step < T; ++step) {
    const size_t base = ((size_t)(b * T + step)) * Cc + h * D;
    if (t < 64) {
      qs[t] = q[base + t];
      ks[t] = k[base + t];
      vs[t] = v[base + t];
    }
    __syncthreads();

    if (t < 32) {
      float s = ks[t] * ks[t] + ks[t + 32] * ks[t + 32];
#pragma unroll
      for (int off = 16; off; off >>= 1) s += __shfl_xor(s, off, 32);
      if (t == 0) s_rinv = 1.f / fmaxf(sqrtf(s), 1e-12f);
    }
    __syncthreads();
    const float rinv = s_rinv;

    float pq = 0.f, pk = 0.f;
#pragma unroll
    for (int jj = 0; jj < 16; ++jj) {
      const float kn = ks[jb + jj] * rinv;
      pq += m[jj] * qs[jb + jj];
      pk += m[jj] * kn;
    }
    pq += __shfl_xor(pq, 1, 32); pq += __shfl_xor(pq, 2, 32);
    pk += __shfl_xor(pk, 1, 32); pk += __shfl_xor(pk, 2, 32);

    if ((t & 3) == 0) q[base + i] = pq;   // y_t

    const float coef = bcoef * vs[i] - acoef * pk;
#pragma unroll
    for (int jj = 0; jj < 16; ++jj) m[jj] += coef * (ks[jb + jj] * rinv);
    __syncthreads();
  }

  const size_t mbase = ((size_t)(b * H + h)) * D * D + (size_t)i * D + jb;
#pragma unroll
  for (int jj = 0; jj < 16; ++jj) M_out[mbase + jj] = m[jj];
}

// ---------------------------------------------------------------------------
// Orchestration. Inputs: x, Wq, Wk, Wv, Wo, alpha_raw, beta_raw (f32).
// Output: concat( y[B,T,C], M_final[B,H,D,D] ), f32.
// Workspace (~120 MB): q,k,v f32 planes; x_h/y_h f16 plane; 4x Wt f16.
// ---------------------------------------------------------------------------
extern "C" void kernel_launch(void* const* d_in, const int* in_sizes, int n_in,
                              void* d_out, int out_size, void* d_ws,
                              size_t ws_size, hipStream_t stream)
{
  (void)in_sizes; (void)n_in; (void)out_size; (void)ws_size;

  const float* x  = (const float*)d_in[0];
  const float* Wq = (const float*)d_in[1];
  const float* Wk = (const float*)d_in[2];
  const float* Wv = (const float*)d_in[3];
  const float* Wo = (const float*)d_in[4];
  const float* alpha_raw = (const float*)d_in[5];
  const float* beta_raw  = (const float*)d_in[6];

  const int Bsz = 4, T = 2048, Cc = 1024, H = 16, D = 64;
  const int Mrows = Bsz * T;                 // 8192
  const size_t plane = (size_t)Mrows * Cc;   // 8,388,608 elems
  const size_t wsz   = (size_t)Cc * Cc;      // 1,048,576 elems

  float* q = (float*)d_ws;
  float* k = q + plane;
  float* v = k + plane;
  _Float16* xh  = (_Float16*)(v + plane);    // reused as y_h after recurrence
  _Float16* wqt = xh + plane;
  _Float16* wkt = wqt + wsz;
  _Float16* wvt = wkt + wsz;
  _Float16* wot = wvt + wsz;

  float* y       = (float*)d_out;            // [B*T, C]
  float* M_final = y + plane;                // [B,H,D,D]

  const int n4 = (int)(plane / 4);
  hipLaunchKernelGGL(f32_to_f16_vec, dim3(n4 / 256), dim3(256), 0, stream,
                     x, xh, n4);

  dim3 tgrid(Cc / 32, Cc / 32), tblk(32, 8);
  hipLaunchKernelGGL(transpose_f32_to_f16, tgrid, tblk, 0, stream, Wq, wqt, Cc, Cc);
  hipLaunchKernelGGL(transpose_f32_to_f16, tgrid, tblk, 0, stream, Wk, wkt, Cc, Cc);
  hipLaunchKernelGGL(transpose_f32_to_f16, tgrid, tblk, 0, stream, Wv, wvt, Cc, Cc);
  hipLaunchKernelGGL(transpose_f32_to_f16, tgrid, tblk, 0, stream, Wo, wot, Cc, Cc);

  dim3 ggrid(Mrows / TILE_M, Cc / TILE_N), gblk(256);
  hipLaunchKernelGGL(gemm_f16_wmma, ggrid, gblk, 0, stream, xh, wqt, q, Mrows, Cc, Cc);
  hipLaunchKernelGGL(gemm_f16_wmma, ggrid, gblk, 0, stream, xh, wkt, k, Mrows, Cc, Cc);
  hipLaunchKernelGGL(gemm_f16_wmma, ggrid, gblk, 0, stream, xh, wvt, v, Mrows, Cc, Cc);

  hipLaunchKernelGGL(titans_recurrence, dim3(Bsz * H), dim3(256), 0, stream,
                     q, k, v, alpha_raw, beta_raw, M_final, Bsz, T, H, D);

  hipLaunchKernelGGL(f32_to_f16_vec, dim3(n4 / 256), dim3(256), 0, stream,
                     q, xh, n4);  // y_pre -> f16 (reuse xh)

  hipLaunchKernelGGL(gemm_f16_wmma, ggrid, gblk, 0, stream, xh, wot, y, Mrows, Cc, Cc);
}
